// FourierAttention_65755949302278
// MI455X (gfx1250) — matrix operand order repaired and checked by reading, not compiled, not run on hardware
//
#include <hip/hip_runtime.h>
#include <stdint.h>
#include <stddef.h>

// ---------------------------------------------------------------------------
// Types for WMMA bf16 (gfx1250: V_WMMA_F32_16X16X32_BF16, wave32)
// ---------------------------------------------------------------------------
typedef __bf16 bf16_t;
typedef bf16_t   v16bf __attribute__((ext_vector_type(16)));
typedef float    v8f   __attribute__((ext_vector_type(8)));
typedef uint32_t u32x4 __attribute__((ext_vector_type(4)));

union FragBF { v16bf v; uint32_t u[8]; };
static_assert(sizeof(FragBF) == 32, "frag must be 8 VGPRs");

__device__ __forceinline__ uint16_t f32_to_bf16_rne(float f) {
  uint32_t u = __float_as_uint(f);
  u += 0x7fffu + ((u >> 16) & 1u);   // round-to-nearest-even
  return (uint16_t)(u >> 16);
}

// ---------------------------------------------------------------------------
// f32 -> bf16 conversion kernel
// ---------------------------------------------------------------------------
__global__ __launch_bounds__(256)
void cvt_bf16_kernel(const float* __restrict__ in, uint16_t* __restrict__ out, int n) {
  int i = blockIdx.x * 256 + threadIdx.x;
  if (i < n) out[i] = f32_to_bf16_rne(in[i]);
}

// ---------------------------------------------------------------------------
// bf16 WMMA GEMM:  Cout[M,Nout] = A[M,K] * W[Nout,K]^T (+ bias)
// A, W row-major bf16 (uint16_t bit patterns), Cout f32.
// Workgroup: 256 threads = 8 waves. Block tile 128x64, K-slab 64 (2 WMMA
// K-steps per LDS stage -> 8 WMMAs per wave per barrier pair).
//
// A tile is staged with GLOBAL_LOAD_ASYNC_TO_LDS_B128 (ASYNCcnt), fenced with
// s_wait_asynccnt 0 before the barrier. Rows beyond M skip the async issue;
// stale LDS in those rows only feeds never-stored OOB output rows.
//
// Fragment layouts (CDNA5 ISA 7.12.2, wave32):
//   A 16x32 bf16 : lane -> m = lane&15, half = lane>>4 selects K sub-blocks;
//                  per lane 16 values: k = (e<8?0:16) + half*8 + (e&7)
//   B 32x16 bf16 : lane -> n = lane&15, half = lane>>4; per lane 16 values: k as for A
//   C/D 16x16 f32: VGPR r, lane L: M = r + 8*(L>>4), N = L&15
// ---------------------------------------------------------------------------
#define GT_M 128
#define GT_N 64
#define GT_K 64

__global__ __launch_bounds__(256)
void gemm_bf16_wmma(const uint16_t* __restrict__ A,
                    const uint16_t* __restrict__ W,
                    float* __restrict__ Cout,
                    const float* __restrict__ bias,
                    int M, int K, int Nout) {
  __shared__ uint16_t lA[GT_M * GT_K];  // [m][k], 16 KB
  __shared__ uint16_t lB[GT_K * GT_N];  // [k][n], 8 KB (transposed on stage-in)

  const int tid  = threadIdx.x;
  const int wave = tid >> 5;
  const int lane = tid & 31;
  const int m0 = blockIdx.x * GT_M;
  const int n0 = blockIdx.y * GT_N;
  const int wm = wave & 3;   // 0..3 -> 32-row band
  const int wn = wave >> 2;  // 0..1 -> 32-col band
  const int half = lane >> 4;
  const int mrow = lane & 15;

  // LDS byte offset of lA (low 32 bits of the generic/flat address)
  const uint32_t lA_base = (uint32_t)(uintptr_t)(void*)&lA[0];

  v8f acc[2][2];
  #pragma unroll
  for (int i = 0; i < 2; ++i)
    #pragma unroll
    for (int j = 0; j < 2; ++j)
      acc[i][j] = (v8f)(0.0f);

  for (int k0 = 0; k0 < K; k0 += GT_K) {
    // ---- stage A tile async: 128 rows x 64 k = 1024 x 16B chunks, 4/thread
    #pragma unroll
    for (int i = 0; i < (GT_M * GT_K / 8) / 256; ++i) {
      int c  = tid + 256 * i;
      int r  = c >> 3;             // GT_K/8 = 8 chunks per row
      int kk = (c & 7) << 3;
      int gm = m0 + r;
      if (gm < M) {
        uint64_t ga = (uint64_t)(uintptr_t)(A + (size_t)gm * K + k0 + kk);
        uint32_t la = lA_base + (uint32_t)((r * GT_K + kk) * 2);
        asm volatile("global_load_async_to_lds_b128 %0, %1, off"
                     :: "v"(la), "v"(ga) : "memory");
      }
    }
    // ---- stage B tile transposed: lB[k][n] = W[n0+n][k0+k]; 2 chunks/thread
    #pragma unroll
    for (int i = 0; i < (GT_N * GT_K / 8) / 256; ++i) {
      int c  = tid + 256 * i;
      int n  = c >> 3;
      int kk = (c & 7) << 3;
      const uint16_t* src = W + (size_t)(n0 + n) * K + k0 + kk;
      if (k0 + GT_K < K)  // prefetch next K-slab of W -> global_prefetch_b8
        __builtin_prefetch((const void*)(src + GT_K), 0, 0);
      #pragma unroll
      for (int j = 0; j < 8; ++j) lB[(kk + j) * GT_N + n] = src[j];
    }
    asm volatile("s_wait_asynccnt 0x0" ::: "memory");
    __syncthreads();

    // ---- two K-substeps of 32; 4 WMMAs each
    #pragma unroll
    for (int ks = 0; ks < 2; ++ks) {
      const int kk0 = ks * 32;
      FragBF fa[2], fb2[2];
      #pragma unroll
      for (int ti = 0; ti < 2; ++ti) {
        const uint32_t* pa =
            (const uint32_t*)(lA + (32 * wm + 16 * ti + mrow) * GT_K + kk0);
        #pragma unroll
        for (int j = 0; j < 4; ++j) {
          fa[ti].u[j]     = pa[half * 4 + j];       // k = kk0 + half*8 + 0..7
          fa[ti].u[4 + j] = pa[8 + half * 4 + j];   // k = kk0 + 16 + half*8 + 0..7
        }
      }
      #pragma unroll
      for (int tj = 0; tj < 2; ++tj) {
        const uint16_t* pb =
            lB + kk0 * GT_N + 32 * wn + 16 * tj + mrow;
        #pragma unroll
        for (int j = 0; j < 8; ++j) {
          int k = (j < 4 ? 0 : 16) + half * 8 + 2 * (j & 3);
          fb2[tj].u[j] = (uint32_t)pb[k * GT_N] | ((uint32_t)pb[(k + 1) * GT_N] << 16);
        }
      }
      #pragma unroll
      for (int ti = 0; ti < 2; ++ti)
        #pragma unroll
        for (int tj = 0; tj < 2; ++tj)
          acc[ti][tj] = __builtin_amdgcn_wmma_f32_16x16x32_bf16(
              false, fa[ti].v, false, fb2[tj].v, (short)0, acc[ti][tj], false, false);
    }
    __syncthreads();
  }

  // ---- writeback (+bias)
  #pragma unroll
  for (int ti = 0; ti < 2; ++ti) {
    int rowbase = m0 + 32 * wm + 16 * ti + 8 * half;
    #pragma unroll
    for (int tj = 0; tj < 2; ++tj) {
      int col = n0 + 32 * wn + 16 * tj + mrow;
      float bv = bias ? bias[col] : 0.0f;
      #pragma unroll
      for (int r = 0; r < 8; ++r) {
        int row = rowbase + r;
        if (row < M) Cout[(size_t)row * Nout + col] = acc[ti][tj][r] + bv;
      }
    }
  }
}

// ---------------------------------------------------------------------------
// Fused per-(b,h) kernel: qt/kt transform -> 2D DFT over 14x14 token grid ->
// attn = sum_d Re(Qf conj(Kf)) -> softmax over 197 -> scale v -> bf16 outcat.
// One 256-thread workgroup per (b,h); D processed in chunks of 8 channels.
// ---------------------------------------------------------------------------
#define AT_P   14
#define AT_N   197
#define AT_D   64
#define AT_DC  8    // channels per chunk (8 chunks)

__global__ __launch_bounds__(256)
void fourier_attn_kernel(const float* __restrict__ qkv,   // [B*197, 2304]
                         const float* __restrict__ wq,    // [12,64,64]
                         const float* __restrict__ wk,    // [12,64,64]
                         uint16_t* __restrict__ outcat) { // [B*197, 768] bf16
  const int b = blockIdx.x / 12;
  const int h = blockIdx.x % 12;
  const int tid = threadIdx.x;

  __shared__ float Ct[AT_P * AT_P], St[AT_P * AT_P];        // DFT twiddles
  __shared__ float qt[AT_N * AT_DC], kt[AT_N * AT_DC];      // transformed chunk
  __shared__ float Gqr[AT_P * AT_P * AT_DC], Gqi[AT_P * AT_P * AT_DC];
  __shared__ float Gkr[AT_P * AT_P * AT_DC], Gki[AT_P * AT_P * AT_DC];
  __shared__ float attn[AT_N];
  __shared__ float red[256];

  if (tid < AT_N) attn[tid] = 0.0f;
  if (tid < AT_P * AT_P) {
    int u = tid / AT_P, x = tid % AT_P;
    float th = (2.0f * 3.14159265358979323846f / (float)AT_P) * (float)(u * x);
    Ct[tid] = __cosf(th);
    St[tid] = __sinf(th);
  }
  __syncthreads();

  const float* qbase = qkv + ((size_t)b * AT_N) * 2304 + h * 64;
  const float* kbase = qbase + 768;
  const float* vbase = qbase + 1536;

  for (int c0 = 0; c0 < AT_D; c0 += AT_DC) {
    // ---- qt/kt: [197, AT_DC] = q/k [197,64] @ wq/wk[h][:, c0:c0+AT_DC]
    for (int i = tid; i < AT_N * AT_DC; i += 256) {
      int n = i / AT_DC, e = i % AT_DC;
      const float* qrow = qbase + (size_t)n * 2304;
      const float* krow = kbase + (size_t)n * 2304;
      float sq = 0.0f, sk = 0.0f;
      #pragma unroll 8
      for (int d = 0; d < AT_D; ++d) {
        float wqv = wq[(size_t)(h * 64 + d) * 64 + c0 + e];
        float wkv = wk[(size_t)(h * 64 + d) * 64 + c0 + e];
        sq = fmaf(qrow[d], wqv, sq);
        sk = fmaf(krow[d], wkv, sk);
      }
      qt[i] = sq; kt[i] = sk;
    }
    __syncthreads();

    // ---- DFT stage 1 over x:  G[u,y,e] = sum_x sp[x,y,e] e^{-2pi i u x / 14}
    // spatial token (x,y) -> token index 1 + x*14 + y
    for (int i = tid; i < AT_P * AT_P * AT_DC; i += 256) {
      int e = i % AT_DC;
      int uy = i / AT_DC;
      int u = uy / AT_P, y = uy % AT_P;
      float qr = 0.0f, qi = 0.0f, kr = 0.0f, ki = 0.0f;
      #pragma unroll
      for (int x = 0; x < AT_P; ++x) {
        float cc = Ct[u * AT_P + x], ss = St[u * AT_P + x];
        float qv = qt[(1 + x * AT_P + y) * AT_DC + e];
        float kv = kt[(1 + x * AT_P + y) * AT_DC + e];
        qr = fmaf(cc, qv, qr);  qi = fmaf(-ss, qv, qi);
        kr = fmaf(cc, kv, kr);  ki = fmaf(-ss, kv, ki);
      }
      Gqr[i] = qr; Gqi[i] = qi; Gkr[i] = kr; Gki[i] = ki;
    }
    __syncthreads();

    // ---- DFT stage 2 over y + accumulate attn[(u,v)] += sum_e Re(Yq conj Yk)
    if (tid < AT_P * AT_P) {
      int u = tid / AT_P, v = tid % AT_P;
      float s = 0.0f;
      #pragma unroll
      for (int e = 0; e < AT_DC; ++e) {
        float yqr = 0.0f, yqi = 0.0f, ykr = 0.0f, yki = 0.0f;
        #pragma unroll
        for (int y = 0; y < AT_P; ++y) {
          float cc = Ct[v * AT_P + y], ss = St[v * AT_P + y];
          float gqr = Gqr[(u * AT_P + y) * AT_DC + e];
          float gqi = Gqi[(u * AT_P + y) * AT_DC + e];
          float gkr = Gkr[(u * AT_P + y) * AT_DC + e];
          float gki = Gki[(u * AT_P + y) * AT_DC + e];
          yqr = fmaf(gqr, cc, fmaf(gqi, ss, yqr));
          yqi = fmaf(gqi, cc, fmaf(-gqr, ss, yqi));
          ykr = fmaf(gkr, cc, fmaf(gki, ss, ykr));
          yki = fmaf(gki, cc, fmaf(-gkr, ss, yki));
        }
        s = fmaf(yqr, ykr, fmaf(yqi, yki, s));
      }
      attn[1 + tid] += s;
    }
    if (tid == 255) {  // cls token: real passthrough -> plain dot
      float s = 0.0f;
      #pragma unroll
      for (int e = 0; e < AT_DC; ++e) s = fmaf(qt[e], kt[e], s);
      attn[0] += s;
    }
    __syncthreads();
  }

  // ---- softmax over attn[0..196]
  float lm = -1e30f;
  for (int i = tid; i < AT_N; i += 256) lm = fmaxf(lm, attn[i]);
  red[tid] = lm; __syncthreads();
  #pragma unroll
  for (int s = 128; s > 0; s >>= 1) {
    if (tid < s) red[tid] = fmaxf(red[tid], red[tid + s]);
    __syncthreads();
  }
  float mx = red[0];
  __syncthreads();
  float ls = 0.0f;
  for (int i = tid; i < AT_N; i += 256) {
    float e = __expf(attn[i] - mx);
    attn[i] = e;
    ls += e;
  }
  red[tid] = ls; __syncthreads();
  #pragma unroll
  for (int s = 128; s > 0; s >>= 1) {
    if (tid < s) red[tid] += red[tid + s];
    __syncthreads();
  }
  float inv = 1.0f / red[0];
  __syncthreads();

  // ---- out[b,n,h*64+d] = bf16( softmax(attn)[n] * v[n,d] )
  for (int i = tid; i < AT_N * AT_D; i += 256) {
    int n = i >> 6, d = i & 63;
    float a  = attn[n] * inv;
    float vv = vbase[(size_t)n * 2304 + d];
    outcat[((size_t)(b * AT_N + n)) * 768 + h * 64 + d] = f32_to_bf16_rne(a * vv);
  }
}

// ---------------------------------------------------------------------------
// Host launcher
// ---------------------------------------------------------------------------
extern "C" void kernel_launch(void* const* d_in, const int* in_sizes, int n_in,
                              void* d_out, int out_size, void* d_ws, size_t ws_size,
                              hipStream_t stream) {
  (void)in_sizes; (void)n_in; (void)out_size; (void)ws_size;

  const float* x      = (const float*)d_in[0];  // [64,197,768]
  const float* qkv_w  = (const float*)d_in[1];  // [2304,768]
  const float* wq     = (const float*)d_in[2];  // [12,64,64]
  const float* wk     = (const float*)d_in[3];  // [12,64,64]
  const float* proj_w = (const float*)d_in[4];  // [768,768]
  const float* proj_b = (const float*)d_in[5];  // [768]
  float* out = (float*)d_out;                   // [64,197,768] f32

  constexpr size_t M  = 64ull * 197;  // 12608 tokens
  constexpr size_t K  = 768;
  constexpr size_t N3 = 2304;

  // workspace carve-up (all 16B aligned; total ~152 MB)
  uint16_t* xb     = (uint16_t*)d_ws;           // bf16 x        [M,K]
  uint16_t* wqkvb  = xb + M * K;                // bf16 qkv_w    [N3,K]
  uint16_t* wprojb = wqkvb + N3 * K;            // bf16 proj_w   [K,K]
  uint16_t* outcat = wprojb + K * K;            // bf16 attn*v   [M,K]
  float*    qkvf   = (float*)(outcat + M * K);  // f32 qkv       [M,N3]

  // 1) convert operands to bf16
  cvt_bf16_kernel<<<(int)((M * K + 255) / 256), 256, 0, stream>>>(x, xb, (int)(M * K));
  cvt_bf16_kernel<<<(int)((N3 * K + 255) / 256), 256, 0, stream>>>(qkv_w, wqkvb, (int)(N3 * K));
  cvt_bf16_kernel<<<(int)((K * K + 255) / 256), 256, 0, stream>>>(proj_w, wprojb, (int)(K * K));

  // 2) QKV GEMM: qkvf = x @ qkv_w^T   (WMMA bf16 -> f32)
  dim3 g1((unsigned)((M + GT_M - 1) / GT_M), (unsigned)(N3 / GT_N));
  gemm_bf16_wmma<<<g1, 256, 0, stream>>>(xb, wqkvb, qkvf, nullptr, (int)M, (int)K, (int)N3);

  // 3) fused transform + FFT attention + softmax + v-scale
  fourier_attn_kernel<<<64 * 12, 256, 0, stream>>>(qkvf, wq, wk, outcat);

  // 4) output projection: out = outcat @ proj_w^T + proj_b   (WMMA bf16 -> f32)
  dim3 g2((unsigned)((M + GT_M - 1) / GT_M), (unsigned)(K / GT_N));
  gemm_bf16_wmma<<<g2, 256, 0, stream>>>(outcat, wprojb, out, proj_b, (int)M, (int)K, (int)K);
}
